// SwiGLUMoETriton_15925738733697
// MI455X (gfx1250) — compile-verified, matrix operand order checked by reading, not run on hardware
//
#include <hip/hip_runtime.h>

// ---------------------------------------------------------------------------
// SwiGLU MoE, grouped per-expert GEMMs on CDNA5 (gfx1250) wave32 WMMA.
// Pipeline: zero(out) -> cvtT (fc11,fc12,fc2 -> bf16, TRANSPOSED [N][K] ws) ->
//           GEMM1 (gather x rows, x@fc11 & x@fc12, SwiGLU -> h bf16 ws) ->
//           GEMM2 (h@fc2 * score, atomic scatter-add into out).
// Tiles are staged into LDS with GLOBAL_LOAD_ASYNC_TO_LDS_B128 (ASYNCcnt) and
// double-buffered: tile k+1 copies fly while tile k feeds the WMMA pipe.
// ---------------------------------------------------------------------------

typedef __attribute__((ext_vector_type(16))) __bf16 v16bf;
typedef __attribute__((ext_vector_type(8)))  float  v8f;

#define T_TOK  8192
#define DMODEL 2048
#define DINT   1024
#define DOUT   2048
#define NEXP   8
#define CAP    2048
#define ATOT   (NEXP * CAP)

#define BM 128   // macrotile rows (assignments)
#define BN 64    // macrotile cols
#define KC 64    // K chunk (2 WMMA K-steps of 32)
#define LDA 72   // LDS stride (bf16 elems); 144B = 16B multiple
#define LDB 72

__device__ __forceinline__ unsigned short f2bf(float f) {
  unsigned int u = __float_as_uint(f);
  u += 0x7FFFu + ((u >> 16) & 1u);     // round-to-nearest-even
  return (unsigned short)(u >> 16);
}

__device__ __forceinline__ unsigned pack_bf2(float a, float b) {
  return (unsigned)f2bf(a) | ((unsigned)f2bf(b) << 16);
}

__device__ __forceinline__ v8f vzero8() {
  v8f r;
#pragma unroll
  for (int i = 0; i < 8; ++i) r[i] = 0.0f;
  return r;
}

// Fragment loader for 16-bit A (16x32) per ISA table; B mirrored, reading from
// an LDS tile stored [N][K].  pair p: K = (p<4 ? 2p : 2p+8) + 8*(lane>=16)
// -> two contiguous 16B runs => compiler emits 2x ds_load_b128.
__device__ __forceinline__ v16bf load_frag(const unsigned short* base, int koff) {
  union { v16bf v; unsigned int u[8]; } f;
#pragma unroll
  for (int p = 0; p < 8; ++p) {
    const int k = (p < 4 ? 2 * p : 2 * p + 8) + koff;
    f.u[p] = *(const unsigned int*)(base + k);
  }
  return f.v;
}

__device__ __forceinline__ v8f wmma_bf16(v16bf a, v16bf b, v8f c) {
  return __builtin_amdgcn_wmma_f32_16x16x32_bf16(
      /*neg_a=*/false, a, /*neg_b=*/false, b,
      /*c_mod=*/(short)0, c, /*reuse_a=*/false, /*reuse_b=*/false);
}

// Async global->LDS 128b copy (GVS mode: saddr + 32b vaddr byte offset).
__device__ __forceinline__ void async_copy_b128(unsigned lds_addr, unsigned voff,
                                                const void* saddr) {
  asm volatile("global_load_async_to_lds_b128 %0, %1, %2 offset:0"
               :: "v"(lds_addr), "v"(voff), "s"(saddr) : "memory");
}

// ---------------------------------------------------------------------------
__global__ void zero_kernel(float4* __restrict__ p, int n4) {
  int i = blockIdx.x * blockDim.x + threadIdx.x;
  int stride = gridDim.x * blockDim.x;
  for (; i < n4; i += stride) p[i] = make_float4(0.f, 0.f, 0.f, 0.f);
}

// fp32 [E][K][N] -> bf16 [E][N][K] (transpose within each expert).
__global__ __launch_bounds__(256) void cvtT_kernel(
    const float* __restrict__ src, unsigned short* __restrict__ dst,
    int K, int N) {
  __shared__ float tile[32][33];
  const int e  = blockIdx.z;
  const int k0 = blockIdx.y * 32;
  const int n0 = blockIdx.x * 32;
  const float* s = src + (size_t)e * K * N;
  unsigned short* d = dst + (size_t)e * K * N;
  const int tx = threadIdx.x & 31;
  const int ty = threadIdx.x >> 5;   // 0..7
#pragma unroll
  for (int i = ty; i < 32; i += 8)
    tile[i][tx] = s[(size_t)(k0 + i) * N + n0 + tx];
  __syncthreads();
#pragma unroll
  for (int i = ty; i < 32; i += 8)
    d[(size_t)(n0 + i) * K + k0 + tx] = f2bf(tile[tx][i]);
}

// ---------------------------------------------------------------------------
// GEMM1: for expert e, rows = gathered tokens. g = x@fc11, v = x@fc12,
// h = g*sigmoid(g)*v -> bf16.  Weights already bf16 [N][K]-major in ws.
// Double-buffered LDS; B via async copies (4 async instr/tile/wave).
__global__ __launch_bounds__(256) void moe_gemm1(
    const float* __restrict__ x,
    const unsigned short* __restrict__ w1t,
    const unsigned short* __restrict__ w2t,
    const int* __restrict__ n_tokens,
    unsigned short* __restrict__ h) {
  __shared__ unsigned short As[2 * BM * LDA];
  __shared__ unsigned short Bs1[2 * BN * LDB];
  __shared__ unsigned short Bs2[2 * BN * LDB];
  __shared__ int toks[BM];

  const int e  = blockIdx.z;
  const int m0 = blockIdx.y * BM;
  const int n0 = blockIdx.x * BN;
  const int tid  = threadIdx.x;
  const int wave = tid >> 5;
  const int lane = tid & 31;
  const int l16  = lane & 15;
  const int koff = (lane >> 4) * 8;

  if (tid < BM) toks[tid] = n_tokens[e * CAP + m0 + tid];
  __syncthreads();

  // A-stage assignment (gather + fp32->bf16 via VGPRs)
  const int arow  = tid >> 1;          // 0..127
  const int acol0 = (tid & 1) * 32;    // half-row of KC=64
  const float* xrow = x + (size_t)toks[arow] * DMODEL;

  // B-stage assignment (async copies; 2 ops per thread per matrix)
  const unsigned short* w1e = w1t + (size_t)e * DMODEL * DINT;
  const unsigned short* w2e = w2t + (size_t)e * DMODEL * DINT;
  const unsigned bs1_base = (unsigned)(size_t)&Bs1[0];
  const unsigned bs2_base = (unsigned)(size_t)&Bs2[0];
  int brow[2], bseg[2];
#pragma unroll
  for (int it = 0; it < 2; ++it) {
    const int i = tid + it * 256;      // 0..511
    brow[it] = i >> 3;                 // 0..63 (n within tile)
    bseg[it] = i & 7;                  // 8 x 16B segments per 64-elem row
  }

  auto issue_b = [&](int k0, int buf) {
    const unsigned lbuf = (unsigned)(buf * BN * LDB * 2);
#pragma unroll
    for (int it = 0; it < 2; ++it) {
      const unsigned goff =
          (unsigned)(((n0 + brow[it]) * DMODEL + k0 + bseg[it] * 8) * 2);
      const unsigned loff = lbuf + (unsigned)((brow[it] * LDB + bseg[it] * 8) * 2);
      async_copy_b128(bs1_base + loff, goff, (const void*)w1e);
      async_copy_b128(bs2_base + loff, goff, (const void*)w2e);
    }
  };
  auto store_a = [&](int k0, int buf) {
    unsigned short* dstA = &As[buf * BM * LDA];
#pragma unroll
    for (int j = 0; j < 8; ++j) {
      float4 f = *(const float4*)(xrow + k0 + acol0 + 4 * j);
      *(unsigned int*)&dstA[arow * LDA + acol0 + 4 * j]     = pack_bf2(f.x, f.y);
      *(unsigned int*)&dstA[arow * LDA + acol0 + 4 * j + 2] = pack_bf2(f.z, f.w);
    }
  };

  v8f accg[4], accv[4];
#pragma unroll
  for (int i = 0; i < 4; ++i) { accg[i] = vzero8(); accv[i] = vzero8(); }

  issue_b(0, 0);
  store_a(0, 0);

  const int NIT = DMODEL / KC;        // 32
  for (int it = 0; it < NIT; ++it) {
    const int cur = it & 1;
    if (it + 1 < NIT) {
      // prefetch tile it+1 into the other buffer (safe: end-of-compute
      // barrier of iteration it-1 guarantees all reads of it are done)
      issue_b((it + 1) * KC, cur ^ 1);
      store_a((it + 1) * KC, cur ^ 1);
      asm volatile("s_wait_asynccnt 0x4" ::: "memory");  // tile it complete
    } else {
      asm volatile("s_wait_asynccnt 0x0" ::: "memory");
    }
    __syncthreads();
    const unsigned short* pA  = &As[cur * BM * LDA];
    const unsigned short* pB1 = &Bs1[cur * BN * LDB];
    const unsigned short* pB2 = &Bs2[cur * BN * LDB];
#pragma unroll
    for (int kk = 0; kk < KC; kk += 32) {
      v16bf af = load_frag(&pA[(wave * 16 + l16) * LDA + kk], koff);
      v16bf bf1[4], bf2[4];
#pragma unroll
      for (int nt = 0; nt < 4; ++nt) {
        bf1[nt] = load_frag(&pB1[(nt * 16 + l16) * LDB + kk], koff);
        bf2[nt] = load_frag(&pB2[(nt * 16 + l16) * LDB + kk], koff);
      }
#pragma unroll
      for (int nt = 0; nt < 4; ++nt) {
        accg[nt] = wmma_bf16(af, bf1[nt], accg[nt]);
        accv[nt] = wmma_bf16(af, bf2[nt], accv[nt]);
      }
    }
    __syncthreads();
  }

  // Epilogue: SwiGLU, write h (bf16). C/D layout: VGPR j -> M = j + 8*(lane>=16)
  const int rbase = e * CAP + m0 + wave * 16 + (lane >> 4) * 8;
#pragma unroll
  for (int nt = 0; nt < 4; ++nt) {
    const int col = n0 + nt * 16 + l16;
#pragma unroll
    for (int j = 0; j < 8; ++j) {
      float g  = accg[nt][j];
      float vv = accv[nt][j];
      float hv = (g * vv) / (1.0f + __expf(-g));
      h[(size_t)(rbase + j) * DINT + col] = f2bf(hv);
    }
  }
}

// ---------------------------------------------------------------------------
// GEMM2: o = h@fc2 * score, atomic scatter-add into out[token].
// Both A (h, bf16) and B (fc2^T, bf16) staged via async copies,
// double-buffered (6 async instr/tile/wave).
__global__ __launch_bounds__(256) void moe_gemm2(
    const unsigned short* __restrict__ h,
    const unsigned short* __restrict__ w3t,
    const int* __restrict__ n_tokens,
    const float* __restrict__ n_score,
    float* __restrict__ out) {
  __shared__ unsigned short As[2 * BM * LDA];
  __shared__ unsigned short Bs[2 * BN * LDB];
  __shared__ int   toks[BM];
  __shared__ float scr[BM];

  const int e  = blockIdx.z;
  const int m0 = blockIdx.y * BM;
  const int n0 = blockIdx.x * BN;
  const int tid  = threadIdx.x;
  const int wave = tid >> 5;
  const int lane = tid & 31;
  const int l16  = lane & 15;
  const int koff = (lane >> 4) * 8;

  if (tid < BM) {
    toks[tid] = n_tokens[e * CAP + m0 + tid];
    scr[tid]  = n_score[e * CAP + m0 + tid];
  }

  const unsigned short* w3e = w3t + (size_t)e * DINT * DOUT;
  const unsigned as_base = (unsigned)(size_t)&As[0];
  const unsigned bs_base = (unsigned)(size_t)&Bs[0];

  auto issue_tiles = [&](int k0, int buf) {
    const unsigned abuf = (unsigned)(buf * BM * LDA * 2);
    const unsigned bbuf = (unsigned)(buf * BN * LDB * 2);
    // A tile: 128 rows x 64 bf16 -> 1024 x 16B ops (4 per thread)
#pragma unroll
    for (int it = 0; it < 4; ++it) {
      const int i   = tid + it * 256;  // 0..1023
      const int row = i >> 3;          // 0..127
      const int seg = i & 7;
      const unsigned goff =
          (unsigned)(((e * CAP + m0 + row) * DINT + k0 + seg * 8) * 2);
      async_copy_b128(as_base + abuf + (unsigned)((row * LDA + seg * 8) * 2),
                      goff, (const void*)h);
    }
    // B tile: 64 rows x 64 bf16 -> 512 x 16B ops (2 per thread)
#pragma unroll
    for (int it = 0; it < 2; ++it) {
      const int i   = tid + it * 256;  // 0..511
      const int row = i >> 3;          // 0..63
      const int seg = i & 7;
      const unsigned goff =
          (unsigned)(((n0 + row) * DINT + k0 + seg * 8) * 2);
      async_copy_b128(bs_base + bbuf + (unsigned)((row * LDB + seg * 8) * 2),
                      goff, (const void*)w3e);
    }
  };

  v8f acc[4];
#pragma unroll
  for (int i = 0; i < 4; ++i) acc[i] = vzero8();

  issue_tiles(0, 0);

  const int NIT = DINT / KC;          // 16
  for (int it = 0; it < NIT; ++it) {
    const int cur = it & 1;
    if (it + 1 < NIT) {
      issue_tiles((it + 1) * KC, cur ^ 1);
      asm volatile("s_wait_asynccnt 0x6" ::: "memory");  // tile it complete
    } else {
      asm volatile("s_wait_asynccnt 0x0" ::: "memory");
    }
    __syncthreads();
    const unsigned short* pA = &As[cur * BM * LDA];
    const unsigned short* pB = &Bs[cur * BN * LDB];
#pragma unroll
    for (int kk = 0; kk < KC; kk += 32) {
      v16bf af = load_frag(&pA[(wave * 16 + l16) * LDA + kk], koff);
      v16bf bfr[4];
#pragma unroll
      for (int nt = 0; nt < 4; ++nt)
        bfr[nt] = load_frag(&pB[(nt * 16 + l16) * LDB + kk], koff);
#pragma unroll
      for (int nt = 0; nt < 4; ++nt)
        acc[nt] = wmma_bf16(af, bfr[nt], acc[nt]);
    }
    __syncthreads();
  }

  // Epilogue: scale by routing score, scatter-add (TOPK=2 -> two commutative
  // adds onto a zeroed output => bit-deterministic).
  const int rloc = wave * 16 + (lane >> 4) * 8;
#pragma unroll
  for (int nt = 0; nt < 4; ++nt) {
    const int col = n0 + nt * 16 + l16;
#pragma unroll
    for (int j = 0; j < 8; ++j) {
      const int r = rloc + j;
      float o = acc[nt][j] * scr[r];
      atomicAdd(out + (size_t)toks[r] * DOUT + col, o);
    }
  }
}

// ---------------------------------------------------------------------------
extern "C" void kernel_launch(void* const* d_in, const int* in_sizes, int n_in,
                              void* d_out, int out_size, void* d_ws, size_t ws_size,
                              hipStream_t stream) {
  (void)in_sizes; (void)n_in; (void)ws_size;
  const float* x        = (const float*)d_in[0];
  const float* fc11     = (const float*)d_in[1];
  const float* fc12     = (const float*)d_in[2];
  const float* fc2      = (const float*)d_in[3];
  const int*   n_tokens = (const int*)d_in[4];
  const float* n_score  = (const float*)d_in[5];
  float* out = (float*)d_out;

  const size_t nW1 = (size_t)NEXP * DMODEL * DINT;   // fc11 / fc12 elems
  const size_t nW3 = (size_t)NEXP * DINT * DOUT;     // fc2 elems
  unsigned short* w1b = (unsigned short*)d_ws;       // bf16 [E][DINT][DMODEL]
  unsigned short* w2b = w1b + nW1;                   // bf16 [E][DINT][DMODEL]
  unsigned short* w3b = w2b + nW1;                   // bf16 [E][DOUT][DINT]
  unsigned short* hb  = w3b + nW3;                   // bf16 [ATOT][DINT]

  zero_kernel<<<dim3(2048), dim3(256), 0, stream>>>((float4*)out, out_size / 4);
  cvtT_kernel<<<dim3(DINT / 32, DMODEL / 32, NEXP), dim3(256), 0, stream>>>(
      fc11, w1b, DMODEL, DINT);
  cvtT_kernel<<<dim3(DINT / 32, DMODEL / 32, NEXP), dim3(256), 0, stream>>>(
      fc12, w2b, DMODEL, DINT);
  cvtT_kernel<<<dim3(DOUT / 32, DINT / 32, NEXP), dim3(256), 0, stream>>>(
      fc2, w3b, DINT, DOUT);

  moe_gemm1<<<dim3(DINT / BN, CAP / BM, NEXP), dim3(256), 0, stream>>>(
      x, w1b, w2b, n_tokens, hb);
  moe_gemm2<<<dim3(DOUT / BN, CAP / BM, NEXP), dim3(256), 0, stream>>>(
      hb, w3b, n_tokens, n_score, out);
}
